// SDTPModel_19104014533469
// MI455X (gfx1250) — compile-verified, matrix operand order checked by reading, not run on hardware
//
#include <hip/hip_runtime.h>

// ---------------------------------------------------------------------------
// Model config (matches reference)
// ---------------------------------------------------------------------------
namespace cfg {
constexpr int B  = 2;
constexpr int S0 = 2048;
constexpr int D  = 2048;
constexpr int HH = 16;     // heads
constexpr int DH = 128;    // head dim
constexpr int FF = 8192;
constexpr int L  = 4;
constexpr int V  = 32000;
}

// ---------------------------------------------------------------------------
// CDNA5 WMMA types / helpers (wave32, 16x16x32 bf16 -> f32)
// ---------------------------------------------------------------------------
typedef __attribute__((ext_vector_type(16))) __bf16 v16bf;
typedef __attribute__((ext_vector_type(8)))  __bf16 v8bf;
typedef __attribute__((ext_vector_type(8)))  float  v8f;

union FragBF {           // one 16x32 (A) or 32x16 (B) bf16 fragment per lane
  v16bf v;
  v8bf  h[2];            // two 16-byte halves -> two b128 loads
};

__device__ __forceinline__ v8f wmma_bf16(const v16bf& a, const v16bf& b, v8f c) {
  // 8 args: (neg_a, A, neg_b, B, c_mod, C, reuse_a, reuse_b)
  return __builtin_amdgcn_wmma_f32_16x16x32_bf16(false, a, false, b, (short)0, c,
                                                 false, false);
}

// ---------------------------------------------------------------------------
// GEMM: C[M,N] = A[M,K] @ W[K,N] (+R), fp32 in/out, bf16 WMMA compute.
// Requires M%128==0, N%128==0, K%32==0 (true for all GEMMs here).
// Block: 256 threads (8 waves). Tile 128x128x32. Wave computes 32x64.
// ---------------------------------------------------------------------------
#define BM 128
#define BN 128
#define BK 32

__global__ __launch_bounds__(256) void gemm_bf16_kernel(
    float* __restrict__ C, const float* __restrict__ A,
    const float* __restrict__ W, const float* __restrict__ R,
    int M, int N, int K)
{
  __shared__ __bf16 sA[BM][BK];   // 8 KB, row-major [m][k]
  __shared__ __bf16 sB[BN][BK];   // 8 KB, transposed weight tile [n][k]

  const int tid    = threadIdx.x;
  const int lane   = tid & 31;
  const int wave   = tid >> 5;          // 0..7
  const int wm     = wave & 3;          // 4 wave-rows of 32
  const int wn     = wave >> 2;         // 2 wave-cols of 64
  const int laneN  = lane & 15;
  const int laneHi = lane >> 4;

  const int bm0 = blockIdx.y * BM;
  const int bn0 = blockIdx.x * BN;

  v8f acc[2][4] = {};

  const int arow = tid >> 3;            // A fill: 32 rows/pass
  const int acol = (tid & 7) * 4;
  const int bn4  = (tid & 31) * 4;      // B fill: coalesced along n
  const int bkr  = tid >> 5;            // 8 k rows/pass

  for (int k0 = 0; k0 < K; k0 += BK) {
    if (k0 + BK < K) {                  // hint next panels (global_prefetch_b8)
      __builtin_prefetch(A + (size_t)(bm0 + arow) * K + k0 + BK + acol, 0, 1);
      __builtin_prefetch(W + (size_t)(k0 + BK + bkr) * N + bn0 + bn4, 0, 1);
    }
    // ---- fill A tile (fp32 -> bf16) ----
    #pragma unroll
    for (int rr = 0; rr < 4; ++rr) {
      const int row = arow + rr * 32;
      const float4 f = *(const float4*)(A + (size_t)(bm0 + row) * K + k0 + acol);
      sA[row][acol + 0] = (__bf16)f.x;  sA[row][acol + 1] = (__bf16)f.y;
      sA[row][acol + 2] = (__bf16)f.z;  sA[row][acol + 3] = (__bf16)f.w;
    }
    // ---- fill B tile transposed: sB[n][k] = W[k0+k][bn0+n] ----
    #pragma unroll
    for (int kk2 = 0; kk2 < 4; ++kk2) {
      const int kr = bkr + kk2 * 8;
      const float4 f = *(const float4*)(W + (size_t)(k0 + kr) * N + bn0 + bn4);
      sB[bn4 + 0][kr] = (__bf16)f.x;  sB[bn4 + 1][kr] = (__bf16)f.y;
      sB[bn4 + 2][kr] = (__bf16)f.z;  sB[bn4 + 3][kr] = (__bf16)f.w;
    }
    __syncthreads();

    // ---- load fragments (two b128 LDS loads each, per ISA A/B layout) ----
    FragBF af[2], bf[4];
    #pragma unroll
    for (int i = 0; i < 2; ++i) {
      const int row = wm * 32 + i * 16 + laneN;
      af[i].h[0] = *(const v8bf*)&sA[row][laneHi * 8];
      af[i].h[1] = *(const v8bf*)&sA[row][laneHi * 8 + 16];
    }
    #pragma unroll
    for (int j = 0; j < 4; ++j) {
      const int col = wn * 64 + j * 16 + laneN;
      bf[j].h[0] = *(const v8bf*)&sB[col][laneHi * 8];
      bf[j].h[1] = *(const v8bf*)&sB[col][laneHi * 8 + 16];
    }
    #pragma unroll
    for (int i = 0; i < 2; ++i)
      #pragma unroll
      for (int j = 0; j < 4; ++j)
        acc[i][j] = wmma_bf16(af[i].v, bf[j].v, acc[i][j]);
    __syncthreads();
  }

  // ---- store (C layout: row = base + laneHi*8 + v, col = base + laneN) ----
  #pragma unroll
  for (int i = 0; i < 2; ++i) {
    #pragma unroll
    for (int j = 0; j < 4; ++j) {
      const int col  = bn0 + wn * 64 + j * 16 + laneN;
      const int rowb = bm0 + wm * 32 + i * 16 + laneHi * 8;
      #pragma unroll
      for (int v = 0; v < 8; ++v) {
        const size_t idx = (size_t)(rowb + v) * N + col;
        float val = acc[i][j][v];
        if (R) val += R[idx];
        C[idx] = val;
      }
    }
  }
}

// ---------------------------------------------------------------------------
// Embedding gather: h[row] = embed[ids[row]]
// ---------------------------------------------------------------------------
__global__ __launch_bounds__(256) void embed_kernel(
    float* __restrict__ h, const float* __restrict__ emb,
    const int* __restrict__ ids, int Dd)
{
  const int row = blockIdx.x;
  const int id  = ids[row];
  const float4* s = (const float4*)(emb + (size_t)id * Dd);
  float4*       d = (float4*)(h + (size_t)row * Dd);
  for (int j = threadIdx.x; j < Dd / 4; j += 256) d[j] = s[j];
}

__global__ void copy_mask_kernel(int* __restrict__ dst,
                                 const int* __restrict__ src, int n)
{
  const int i = blockIdx.x * blockDim.x + threadIdx.x;
  if (i < n) dst[i] = src[i];
}

// ---------------------------------------------------------------------------
// RMSNorm: out = x * w * rsqrt(mean(x^2)+eps), one block per row
// ---------------------------------------------------------------------------
__global__ __launch_bounds__(256) void rmsnorm_kernel(
    float* __restrict__ out, const float* __restrict__ in,
    const float* __restrict__ w, int Dd)
{
  __shared__ float red[8];
  const int row = blockIdx.x, tid = threadIdx.x;
  const float* x = in + (size_t)row * Dd;
  float ss = 0.f;
  for (int d = tid; d < Dd; d += 256) { const float v = x[d]; ss += v * v; }
  for (int off = 1; off < 32; off <<= 1) ss += __shfl_xor(ss, off, 32);
  if ((tid & 31) == 0) red[tid >> 5] = ss;
  __syncthreads();
  if (tid == 0) {
    float t = 0.f;
    for (int k = 0; k < 8; ++k) t += red[k];
    red[0] = t;
  }
  __syncthreads();
  const float r = rsqrtf(red[0] / (float)Dd + 1e-6f);
  float* o = out + (size_t)row * Dd;
  for (int d = tid; d < Dd; d += 256) o[d] = x[d] * w[d] * r;
}

// ---------------------------------------------------------------------------
// RoPE + pack to bf16 head-major: Q,K -> [B,H,S,DH]; V -> [B,H,DH,S] (transposed)
// grid = B*H*S, block = 64 (one rotation pair per thread)
// ---------------------------------------------------------------------------
__global__ __launch_bounds__(64) void rope_pack_kernel(
    __bf16* __restrict__ Qo, __bf16* __restrict__ Ko, __bf16* __restrict__ Vt,
    const float* __restrict__ Qi, const float* __restrict__ Ki,
    const float* __restrict__ Vi, int S)
{
  using namespace cfg;
  const int idx = blockIdx.x;
  const int s  = idx % S;
  const int bh = idx / S;
  const int h  = bh % HH;
  const int b  = bh / HH;
  const int d  = threadIdx.x;                       // 0..63

  const size_t inoff = ((size_t)b * S + s) * D + (size_t)h * DH;
  const float inv = __powf(10000.f, -(float)d / 64.f);
  float sn, cs;
  __sincosf((float)s * inv, &sn, &cs);

  const float q1 = Qi[inoff + d], q2 = Qi[inoff + d + 64];
  const float k1 = Ki[inoff + d], k2 = Ki[inoff + d + 64];

  const size_t outoff = (((size_t)b * HH + h) * S + s) * DH;
  Qo[outoff + d]      = (__bf16)(q1 * cs - q2 * sn);
  Qo[outoff + d + 64] = (__bf16)(q1 * sn + q2 * cs);
  Ko[outoff + d]      = (__bf16)(k1 * cs - k2 * sn);
  Ko[outoff + d + 64] = (__bf16)(k1 * sn + k2 * cs);

  const size_t vb = ((size_t)b * HH + h) * (size_t)DH * S;
  Vt[vb + (size_t)d * S + s]        = (__bf16)Vi[inoff + d];
  Vt[vb + (size_t)(d + 64) * S + s] = (__bf16)Vi[inoff + d + 64];
}

// ---------------------------------------------------------------------------
// Flash attention (causal + key mask), WMMA everywhere.
// grid = B*H*(S/128), block = 256 (8 waves); each wave owns a 16-row q tile.
// Q,K: bf16 [B,H,S,DH]; V: bf16 [B,H,DH,S]; O: fp32 [B,S,D] token-major.
// ---------------------------------------------------------------------------
__global__ __launch_bounds__(256) void flash_attn_kernel(
    float* __restrict__ O, const __bf16* __restrict__ Q,
    const __bf16* __restrict__ Kb, const __bf16* __restrict__ Vt,
    const int* __restrict__ mask, int S)
{
  using namespace cfg;
  const int tiles = S / 128;
  const int qblk  = blockIdx.x % tiles;
  const int bh    = blockIdx.x / tiles;
  const int h     = bh % HH;
  const int b     = bh / HH;

  const int lane   = threadIdx.x & 31;
  const int wave   = threadIdx.x >> 5;
  const int laneN  = lane & 15;
  const int laneHi = lane >> 4;
  const int q0     = qblk * 128 + wave * 16;

  const __bf16* qp = Q  + (((size_t)b * HH + h) * S + q0) * DH;
  const __bf16* kp = Kb + (((size_t)b * HH + h) * S) * (size_t)DH;
  const __bf16* vp = Vt + (((size_t)b * HH + h) * DH) * (size_t)S;
  const int* mrow  = mask + (size_t)b * S;

  __shared__ __bf16 pbuf[8][16][32];   // wave-private P tiles (8 KB)

  // q fragments: 4 chunks of K=32 over DH=128
  FragBF qf[4];
  {
    const __bf16* base = qp + (size_t)laneN * DH + laneHi * 8;
    #pragma unroll
    for (int c = 0; c < 4; ++c) {
      qf[c].h[0] = *(const v8bf*)(base + c * 32);
      qf[c].h[1] = *(const v8bf*)(base + c * 32 + 16);
    }
  }

  v8f oacc[8] = {};
  float rmax[8], rsum[8];
  #pragma unroll
  for (int v = 0; v < 8; ++v) { rmax[v] = -3.0e38f; rsum[v] = 0.f; }
  const float scale = 0.08838834764831845f;   // 1/sqrt(128)

  const int ksteps = (q0 + 16 + 31) / 32;     // causal upper bound, 32 keys/step
  for (int ks = 0; ks < ksteps; ++ks) {
    const int kb0 = ks * 32;

    // ---- scores: two 16x16 tiles over 32 keys ----
    v8f s0 = {}, s1 = {};
    #pragma unroll
    for (int c = 0; c < 4; ++c) {
      FragBF b0, b1;
      const __bf16* k0p = kp + (size_t)(kb0 + laneN) * DH + c * 32 + laneHi * 8;
      const __bf16* k1p = k0p + (size_t)16 * DH;
      b0.h[0] = *(const v8bf*)k0p;  b0.h[1] = *(const v8bf*)(k0p + 16);
      b1.h[0] = *(const v8bf*)k1p;  b1.h[1] = *(const v8bf*)(k1p + 16);
      s0 = wmma_bf16(qf[c].v, b0.v, s0);
      s1 = wmma_bf16(qf[c].v, b1.v, s1);
    }

    // ---- online softmax in C layout (per-lane column, 8 rows in VGPRs) ----
    const int key0 = kb0 + laneN, key1 = key0 + 16;
    const bool ok0 = mrow[key0] != 0;
    const bool ok1 = mrow[key1] != 0;
    #pragma unroll
    for (int v = 0; v < 8; ++v) {
      const int qi = q0 + laneHi * 8 + v;
      const float x0 = (ok0 && key0 <= qi) ? s0[v] * scale : -1e9f;
      const float x1 = (ok1 && key1 <= qi) ? s1[v] * scale : -1e9f;
      float mx = fmaxf(x0, x1);
      for (int off = 1; off < 16; off <<= 1) mx = fmaxf(mx, __shfl_xor(mx, off, 32));
      const float nm   = fmaxf(rmax[v], mx);
      const float corr = __expf(rmax[v] - nm);
      const float e0 = __expf(x0 - nm);
      const float e1 = __expf(x1 - nm);
      float ls = e0 + e1;
      for (int off = 1; off < 16; off <<= 1) ls += __shfl_xor(ls, off, 32);
      rsum[v] = rsum[v] * corr + ls;
      rmax[v] = nm;
      #pragma unroll
      for (int d = 0; d < 8; ++d) oacc[d][v] *= corr;
      const int m = laneHi * 8 + v;
      pbuf[wave][m][laneN]      = (__bf16)e0;   // C layout -> A layout via LDS
      pbuf[wave][m][laneN + 16] = (__bf16)e1;
    }

    // ---- P as A fragment (16x32), then O += P @ V over 8 dh tiles ----
    FragBF pf;
    pf.h[0] = *(const v8bf*)&pbuf[wave][laneN][laneHi * 8];
    pf.h[1] = *(const v8bf*)&pbuf[wave][laneN][laneHi * 8 + 16];
    #pragma unroll
    for (int d = 0; d < 8; ++d) {
      FragBF vf;
      const __bf16* vbase = vp + (size_t)(d * 16 + laneN) * S + kb0 + laneHi * 8;
      vf.h[0] = *(const v8bf*)vbase;  vf.h[1] = *(const v8bf*)(vbase + 16);
      oacc[d] = wmma_bf16(pf.v, vf.v, oacc[d]);
    }
  }

  // ---- finalize & store to token-major [B,S,D] ----
  #pragma unroll
  for (int d = 0; d < 8; ++d) {
    const int col = h * DH + d * 16 + laneN;
    #pragma unroll
    for (int v = 0; v < 8; ++v) {
      const int qi = q0 + laneHi * 8 + v;
      O[((size_t)b * S + qi) * D + col] = oacc[d][v] / rsum[v];
    }
  }
}

// ---------------------------------------------------------------------------
// SwiGLU elementwise: act = silu(g) * u (in-place into g is fine)
// ---------------------------------------------------------------------------
__global__ void silu_mul_kernel(float* __restrict__ act,
                                const float* __restrict__ g,
                                const float* __restrict__ u, size_t n)
{
  const size_t i = (size_t)blockIdx.x * blockDim.x + threadIdx.x;
  if (i < n) {
    const float x = g[i];
    act[i] = (x / (1.f + __expf(-x))) * u[i];
  }
}

// ---------------------------------------------------------------------------
// Scorer dot: sc[row] = dot(h[row], w)   (one block per row)
// ---------------------------------------------------------------------------
__global__ __launch_bounds__(256) void score_kernel(
    float* __restrict__ sc, const float* __restrict__ h,
    const float* __restrict__ w, int Dd)
{
  __shared__ float red[8];
  const int row = blockIdx.x, tid = threadIdx.x;
  const float* x = h + (size_t)row * Dd;
  float s = 0.f;
  for (int d = tid; d < Dd; d += 256) s += x[d] * w[d];
  for (int off = 1; off < 32; off <<= 1) s += __shfl_xor(s, off, 32);
  if ((tid & 31) == 0) red[tid >> 5] = s;
  __syncthreads();
  if (tid == 0) {
    float t = 0.f;
    for (int k = 0; k < 8; ++k) t += red[k];
    sc[row] = t;
  }
}

// ---------------------------------------------------------------------------
// Top-k keep (matches jax top_k tie-break: higher score first, then lower idx),
// kept tokens emitted in ascending original index (matches sort(idx)).
// One block (1024 threads) per batch. Writes index map + compacted mask.
// ---------------------------------------------------------------------------
__global__ __launch_bounds__(1024) void prune_kernel(
    int* __restrict__ idxmap, int* __restrict__ mask_new,
    const float* __restrict__ sc, const int* __restrict__ mask_old,
    int Scur, int kk)
{
  __shared__ float s_sc[2048];
  __shared__ unsigned char s_keep[2048];
  const int b = blockIdx.x, tid = threadIdx.x;
  for (int i = tid; i < Scur; i += 1024) s_sc[i] = sc[(size_t)b * Scur + i];
  __syncthreads();
  for (int i = tid; i < Scur; i += 1024) {
    const float si = s_sc[i];
    int rank = 0;
    for (int j = 0; j < Scur; ++j) {
      const float sj = s_sc[j];
      rank += (sj > si) || (sj == si && j < i);
    }
    s_keep[i] = (rank < kk) ? 1 : 0;
  }
  __syncthreads();
  for (int i = tid; i < Scur; i += 1024) {
    if (s_keep[i]) {
      int pos = 0;
      for (int j = 0; j < i; ++j) pos += s_keep[j];
      idxmap[(size_t)b * kk + pos]   = i;
      mask_new[(size_t)b * kk + pos] = mask_old[(size_t)b * Scur + i];
    }
  }
}

__global__ __launch_bounds__(256) void gather_rows_kernel(
    float* __restrict__ dst, const float* __restrict__ src,
    const int* __restrict__ idxmap, int Sold, int kk, int Dd)
{
  const int row = blockIdx.x;            // b*kk + pos
  const int b   = row / kk;
  const int si  = idxmap[row];
  const float4* s = (const float4*)(src + ((size_t)b * Sold + si) * Dd);
  float4*       d = (float4*)(dst + (size_t)row * Dd);
  for (int j = threadIdx.x; j < Dd / 4; j += 256) d[j] = s[j];
}

// ---------------------------------------------------------------------------
// Host orchestration
// ---------------------------------------------------------------------------
extern "C" void kernel_launch(void* const* d_in, const int* in_sizes, int n_in,
                              void* d_out, int out_size, void* d_ws, size_t ws_size,
                              hipStream_t stream)
{
  using namespace cfg;
  (void)in_sizes; (void)n_in; (void)out_size; (void)ws_size;

  const int*   ids     = (const int*)d_in[0];
  const int*   amask   = (const int*)d_in[1];
  const float* embed   = (const float*)d_in[2];
  const float* wq      = (const float*)d_in[3];
  const float* wk      = (const float*)d_in[4];
  const float* wv      = (const float*)d_in[5];
  const float* wo      = (const float*)d_in[6];
  const float* wg      = (const float*)d_in[7];
  const float* wu      = (const float*)d_in[8];
  const float* wd      = (const float*)d_in[9];
  const float* ln1     = (const float*)d_in[10];
  const float* ln2     = (const float*)d_in[11];
  const float* lnf     = (const float*)d_in[12];
  const float* lm_head = (const float*)d_in[13];
  const float* scorer  = (const float*)d_in[14];
  float* out = (float*)d_out;

  // ---- carve workspace (deterministic layout) ----
  char* wsp = (char*)d_ws;
  auto carve = [&](size_t bytes) -> void* {
    void* r = (void*)wsp;
    wsp += (bytes + 255) & ~(size_t)255;
    return r;
  };
  const size_t MAXM = (size_t)B * S0;               // 4096
  float*  hA    = (float*)carve(MAXM * D * 4);
  float*  hB    = (float*)carve(MAXM * D * 4);
  float*  hn    = (float*)carve(MAXM * D * 4);
  float*  qb    = (float*)carve(MAXM * D * 4);
  float*  kb    = (float*)carve(MAXM * D * 4);
  float*  vb    = (float*)carve(MAXM * D * 4);
  float*  obuf  = (float*)carve(MAXM * D * 4);
  __bf16* qbf   = (__bf16*)carve(MAXM * D * 2);
  __bf16* kbf   = (__bf16*)carve(MAXM * D * 2);
  __bf16* vbf   = (__bf16*)carve(MAXM * D * 2);
  float*  gbuf  = (float*)carve(MAXM * FF * 4);
  float*  ubuf  = (float*)carve(MAXM * FF * 4);
  float*  scbuf = (float*)carve(MAXM * 4);
  int*    idxm  = (int*)carve(MAXM * 4);
  int*    maskA = (int*)carve(MAXM * 4);
  int*    maskB = (int*)carve(MAXM * 4);

  int Scur = S0;
  float* hcur = hA;  float* halt = hB;
  int*   mcur = maskA; int* malt = maskB;

  embed_kernel<<<B * Scur, 256, 0, stream>>>(hcur, embed, ids, D);
  copy_mask_kernel<<<(B * S0 + 255) / 256, 256, 0, stream>>>(mcur, amask, B * S0);

  for (int i = 0; i < L; ++i) {
    if (i == 1 || i == 2) {                   // PRUNE_LAYERS
      const int pi = i - 1;
      const int M  = B * Scur;
      const int kk = Scur / 2;
      score_kernel<<<M, 256, 0, stream>>>(scbuf, hcur, scorer + (size_t)pi * 2 * D, D);
      prune_kernel<<<B, 1024, 0, stream>>>(idxm, malt, scbuf, mcur, Scur, kk);
      gather_rows_kernel<<<B * kk, 256, 0, stream>>>(halt, hcur, idxm, Scur, kk, D);
      { float* t = hcur; hcur = halt; halt = t; }
      { int*   t = mcur; mcur = malt; malt = t; }
      Scur = kk;
    }
    const int M = B * Scur;
    const dim3 gDD(D / BN, M / BM), gDF(FF / BN, M / BM);

    // attention
    rmsnorm_kernel<<<M, 256, 0, stream>>>(hn, hcur, ln1 + (size_t)i * D, D);
    gemm_bf16_kernel<<<gDD, 256, 0, stream>>>(qb, hn, wq + (size_t)i * D * D, nullptr, M, D, D);
    gemm_bf16_kernel<<<gDD, 256, 0, stream>>>(kb, hn, wk + (size_t)i * D * D, nullptr, M, D, D);
    gemm_bf16_kernel<<<gDD, 256, 0, stream>>>(vb, hn, wv + (size_t)i * D * D, nullptr, M, D, D);
    rope_pack_kernel<<<B * HH * Scur, 64, 0, stream>>>(qbf, kbf, vbf, qb, kb, vb, Scur);
    flash_attn_kernel<<<B * HH * (Scur / 128), 256, 0, stream>>>(obuf, qbf, kbf, vbf, mcur, Scur);
    gemm_bf16_kernel<<<gDD, 256, 0, stream>>>(hcur, obuf, wo + (size_t)i * D * D, hcur, M, D, D);

    // MLP
    rmsnorm_kernel<<<M, 256, 0, stream>>>(hn, hcur, ln2 + (size_t)i * D, D);
    gemm_bf16_kernel<<<gDF, 256, 0, stream>>>(gbuf, hn, wg + (size_t)i * D * FF, nullptr, M, FF, D);
    gemm_bf16_kernel<<<gDF, 256, 0, stream>>>(ubuf, hn, wu + (size_t)i * D * FF, nullptr, M, FF, D);
    const size_t nel = (size_t)M * FF;
    silu_mul_kernel<<<(unsigned)((nel + 255) / 256), 256, 0, stream>>>(gbuf, gbuf, ubuf, nel);
    gemm_bf16_kernel<<<gDD, 256, 0, stream>>>(hcur, gbuf, wd + (size_t)i * FF * D, hcur, M, D, FF);
  }

  // final norm + LM head (writes d_out directly: (B, 512, V))
  const int M = B * Scur;
  rmsnorm_kernel<<<M, 256, 0, stream>>>(hn, hcur, lnf, D);
  gemm_bf16_kernel<<<dim3(V / BN, M / BM), 256, 0, stream>>>(out, hn, lm_head, nullptr, M, V, D);
}
